// _PointnetSAModuleFSBase_31576599560751
// MI455X (gfx1250) — matrix operand order repaired and checked by reading, not compiled, not run on hardware
//
#include <hip/hip_runtime.h>

// ---------------------------------------------------------------------------
// PointNet++ SA module for MI455X (gfx1250, wave32).
// Pipeline: FPS -> ball query (2 radii) -> grouped MLP (WMMA f16) + maxpool
//           -> aggregation GEMM (WMMA f16) + ReLU -> confidence head.
// LDS staging tiles are column-major so WMMA operand builds are b128 DS ops.
// ---------------------------------------------------------------------------

typedef __attribute__((ext_vector_type(16))) _Float16 v16h;
typedef __attribute__((ext_vector_type(8)))  _Float16 v8h;
typedef __attribute__((ext_vector_type(8)))  float    v8f;

#define BB      4
#define NN      16384
#define CC      16
#define NPT     4096
#define AGG     128

// Wave-local LDS fence: DS ops from one wave are in-order; wait for them to
// land before cross-lane consumption. "memory" clobber stops reordering.
#define LDS_FENCE() asm volatile("s_wait_dscnt 0" ::: "memory")

union V16H { v16h v; v8h q[2]; _Float16 h[16]; };
union V8H  { v8h  v; _Float16 h[8]; };

static __device__ __forceinline__ v8f wmma_f16(v16h a, v16h b, v8f c) {
  return __builtin_amdgcn_wmma_f32_16x16x32_f16(false, a, false, b, (short)0, c,
                                                false, false);
}

// A-operand element->K mapping for 16-bit 16x32 A tiles (lane half hf):
//   K = e + 16*hf (contiguous, matches B operand builds)
static __device__ __forceinline__ int a_kmap(int e, int hf) {
  return e + 16 * hf;
}

// ---------------------------------------------------------------------------
// Kernel 1: furthest point sampling, one block per batch.
// ---------------------------------------------------------------------------
__global__ __launch_bounds__(1024) void fps_kernel(
    const float* __restrict__ xyz, int* __restrict__ sidx,
    float* __restrict__ newxyz) {
  const int b   = blockIdx.x;
  const int tid = threadIdx.x;
  const float* p = xyz + (size_t)b * NN * 3;

  float px[16], py[16], pz[16], md[16];
#pragma unroll
  for (int k = 0; k < 16; ++k) {
    int i = tid + k * 1024;
    px[k] = p[i * 3 + 0];
    py[k] = p[i * 3 + 1];
    pz[k] = p[i * 3 + 2];
    md[k] = 1e10f;
  }

  __shared__ float s_val[32];
  __shared__ int   s_idx[32];
  __shared__ int   s_cur;

  int cur = 0;
  for (int it = 0; it < NPT; ++it) {
    if (tid == 0) {
      sidx[b * NPT + it] = cur;
      newxyz[(size_t)(b * NPT + it) * 3 + 0] = p[cur * 3 + 0];
      newxyz[(size_t)(b * NPT + it) * 3 + 1] = p[cur * 3 + 1];
      newxyz[(size_t)(b * NPT + it) * 3 + 2] = p[cur * 3 + 2];
    }
    if (it == NPT - 1) break;

    float cx = p[cur * 3 + 0], cy = p[cur * 3 + 1], cz = p[cur * 3 + 2];
    float best = -1.0f;
    int   bi   = 0x7fffffff;
#pragma unroll
    for (int k = 0; k < 16; ++k) {
      float dx = px[k] - cx, dy = py[k] - cy, dz = pz[k] - cz;
      float d  = dx * dx + dy * dy + dz * dz;
      md[k]    = fminf(md[k], d);
      if (md[k] > best) { best = md[k]; bi = tid + k * 1024; }
    }
    // wave32 argmax reduction (ties -> smaller index)
#pragma unroll
    for (int off = 16; off >= 1; off >>= 1) {
      float ov = __shfl_xor(best, off);
      int   oi = __shfl_xor(bi, off);
      if (ov > best || (ov == best && oi < bi)) { best = ov; bi = oi; }
    }
    int lane = tid & 31, wv = tid >> 5;
    if (lane == 0) { s_val[wv] = best; s_idx[wv] = bi; }
    __syncthreads();
    if (wv == 0) {
      best = s_val[lane];
      bi   = s_idx[lane];
#pragma unroll
      for (int off = 16; off >= 1; off >>= 1) {
        float ov = __shfl_xor(best, off);
        int   oi = __shfl_xor(bi, off);
        if (ov > best || (ov == best && oi < bi)) { best = ov; bi = oi; }
      }
      if (lane == 0) s_cur = bi;
    }
    __syncthreads();
    cur = s_cur;
  }
}

// ---------------------------------------------------------------------------
// Kernel 2: dual-radius ball query, one wave per (batch, sample) group.
// Ordered first-k selection via ballot + prefix popcount, early exit.
// ---------------------------------------------------------------------------
__global__ __launch_bounds__(256) void ballq_kernel(
    const float* __restrict__ xyz, const float* __restrict__ newxyz,
    int* __restrict__ idx0, int* __restrict__ idx1, int* __restrict__ cnt) {
  const int wave = (blockIdx.x * blockDim.x + threadIdx.x) >> 5;
  if (wave >= BB * NPT) return;
  const int lane = threadIdx.x & 31;
  const int b = wave / NPT;
  const int j = wave % NPT;

  const float cx = newxyz[(size_t)(b * NPT + j) * 3 + 0];
  const float cy = newxyz[(size_t)(b * NPT + j) * 3 + 1];
  const float cz = newxyz[(size_t)(b * NPT + j) * 3 + 2];
  const float* p = xyz + (size_t)b * NN * 3;

  int* o0 = idx0 + (size_t)wave * 16;
  int* o1 = idx1 + (size_t)wave * 32;

  int c0 = 0, c1 = 0, f0 = NN - 1, f1 = NN - 1;
  const unsigned lm = (lane == 0) ? 0u : ((1u << lane) - 1u);

  for (int base = 0; base < NN && (c0 < 16 || c1 < 32); base += 32) {
    if (base + 256 < NN)  // stream ahead: emits global_prefetch_b8
      __builtin_prefetch(&p[(base + 256) * 3 + lane], 0, 0);
    int   i  = base + lane;
    float dx = p[i * 3 + 0] - cx;
    float dy = p[i * 3 + 1] - cy;
    float dz = p[i * 3 + 2] - cz;
    float d2 = dx * dx + dy * dy + dz * dz;
    bool  w0 = d2 < 0.25f;
    bool  w1 = d2 < 1.0f;
    unsigned m0 = (unsigned)__ballot(w0);
    unsigned m1 = (unsigned)__ballot(w1);
    if (c0 == 0 && m0) f0 = __shfl(i, __ffs((int)m0) - 1);
    if (c1 == 0 && m1) f1 = __shfl(i, __ffs((int)m1) - 1);
    int s0 = c0 + __popc(m0 & lm);
    if (w0 && s0 < 16) o0[s0] = i;
    int s1 = c1 + __popc(m1 & lm);
    if (w1 && s1 < 32) o1[s1] = i;
    c0 += __popc(m0);
    c1 += __popc(m1);
  }
  int m0c = c0 < 16 ? c0 : 16;
  int m1c = c1 < 32 ? c1 : 32;
  if (lane < 16 && lane >= m0c) o0[lane] = f0;
  if (lane >= m1c)              o1[lane] = f1;
  if (lane == 0) { cnt[wave * 2 + 0] = c0; cnt[wave * 2 + 1] = c1; }
}

// ---------------------------------------------------------------------------
// Kernel 3: gather + 3-layer MLP (19->32->32->64) + neighbor max-pool.
// One wave per sample; WMMA f32_16x16x32_f16; weights live in A-layout regs.
// LDS tile is column-major X[col][k] so operand builds are b128 DS ops.
// ---------------------------------------------------------------------------
__global__ __launch_bounds__(256) void mlp_kernel(
    const float* __restrict__ xyz, const float* __restrict__ feats,
    const float* __restrict__ newxyz, const int* __restrict__ idxb,
    const int* __restrict__ cnt, int g, int ns,
    const float* __restrict__ w1, const float* __restrict__ b1,
    const float* __restrict__ w2, const float* __restrict__ b2,
    const float* __restrict__ w3, const float* __restrict__ b3,
    float* __restrict__ outfeat, int choff) {
  const int wave = (blockIdx.x * blockDim.x + threadIdx.x) >> 5;
  if (wave >= BB * NPT) return;
  const int lane = threadIdx.x & 31;
  const int hf   = lane >> 4;
  const int l15  = lane & 15;
  const int wlb  = (threadIdx.x >> 5) & 7;
  const int b    = wave / NPT;
  const int j    = wave % NPT;

  // Preload weights into WMMA A-operand layout (f32 -> f16), once per wave.
  V16H A1[2], A2[2], A3[4];
#pragma unroll
  for (int e = 0; e < 16; ++e) {
    int k = a_kmap(e, hf);
#pragma unroll
    for (int t = 0; t < 2; ++t)
      A1[t].h[e] = (k < 19) ? (_Float16)w1[(t * 16 + l15) * 19 + k]
                            : (_Float16)0.f;
#pragma unroll
    for (int t = 0; t < 2; ++t)
      A2[t].h[e] = (_Float16)w2[(t * 16 + l15) * 32 + k];
#pragma unroll
    for (int t = 0; t < 4; ++t)
      A3[t].h[e] = (_Float16)w3[(t * 16 + l15) * 32 + k];
  }

  const float cx = newxyz[(size_t)(b * NPT + j) * 3 + 0];
  const float cy = newxyz[(size_t)(b * NPT + j) * 3 + 1];
  const float cz = newxyz[(size_t)(b * NPT + j) * 3 + 2];
  const float* p = xyz + (size_t)b * NN * 3;

  // Column-major staging: xb[wave][col*32 + k], 64B per column (aligned).
  __shared__ _Float16 xb[8][16 * 32];

  float rmax[4][8];
#pragma unroll
  for (int t = 0; t < 4; ++t)
#pragma unroll
    for (int i = 0; i < 8; ++i) rmax[t][i] = 0.0f;  // ReLU output >= 0

  const int ntile = ns >> 4;
  for (int t = 0; t < ntile; ++t) {
    LDS_FENCE();  // WAR vs. previous tile's B-build loads
    if (lane < 16) {
      int id = idxb[(size_t)wave * ns + t * 16 + lane];
      V16H lo, hi;
      lo.h[0] = (_Float16)(p[id * 3 + 0] - cx);
      lo.h[1] = (_Float16)(p[id * 3 + 1] - cy);
      lo.h[2] = (_Float16)(p[id * 3 + 2] - cz);
#pragma unroll
      for (int c = 0; c < 13; ++c)
        lo.h[3 + c] = (_Float16)feats[((size_t)b * CC + c) * NN + id];
#pragma unroll
      for (int c = 13; c < 16; ++c)
        hi.h[c - 13] = (_Float16)feats[((size_t)b * CC + c) * NN + id];
#pragma unroll
      for (int k = 3; k < 16; ++k) hi.h[k] = (_Float16)0.f;
      *(v16h*)&xb[wlb][lane * 32 + 0]  = lo.v;   // 2x ds_store_b128
      *(v16h*)&xb[wlb][lane * 32 + 16] = hi.v;   // 2x ds_store_b128
    }
    LDS_FENCE();

    // Layer 1: 19(pad 32) -> 32
    V16H Bv;
    Bv.v = *(const v16h*)&xb[wlb][l15 * 32 + 16 * hf];  // 2x ds_load_b128
    v8f c0 = {}, c1 = {};
    c0 = wmma_f16(A1[0].v, Bv.v, c0);
    c1 = wmma_f16(A1[1].v, Bv.v, c1);
    LDS_FENCE();
    {
      V8H u0, u1;
#pragma unroll
      for (int i = 0; i < 8; ++i) {
        u0.h[i] = (_Float16)fmaxf(c0[i] + b1[i + 8 * hf], 0.f);
        u1.h[i] = (_Float16)fmaxf(c1[i] + b1[16 + i + 8 * hf], 0.f);
      }
      *(v8h*)&xb[wlb][l15 * 32 + 8 * hf]      = u0.v;  // rows 8hf..8hf+7
      *(v8h*)&xb[wlb][l15 * 32 + 16 + 8 * hf] = u1.v;  // rows 16+8hf..
    }
    LDS_FENCE();

    // Layer 2: 32 -> 32
    Bv.v = *(const v16h*)&xb[wlb][l15 * 32 + 16 * hf];
    v8f d0 = {}, d1 = {};
    d0 = wmma_f16(A2[0].v, Bv.v, d0);
    d1 = wmma_f16(A2[1].v, Bv.v, d1);
    LDS_FENCE();
    {
      V8H u0, u1;
#pragma unroll
      for (int i = 0; i < 8; ++i) {
        u0.h[i] = (_Float16)fmaxf(d0[i] + b2[i + 8 * hf], 0.f);
        u1.h[i] = (_Float16)fmaxf(d1[i] + b2[16 + i + 8 * hf], 0.f);
      }
      *(v8h*)&xb[wlb][l15 * 32 + 8 * hf]      = u0.v;
      *(v8h*)&xb[wlb][l15 * 32 + 16 + 8 * hf] = u1.v;
    }
    LDS_FENCE();

    // Layer 3: 32 -> 64, fold ReLU + running neighbor max
    Bv.v = *(const v16h*)&xb[wlb][l15 * 32 + 16 * hf];
    v8f E0 = {}, E1 = {}, E2 = {}, E3 = {};
    E0 = wmma_f16(A3[0].v, Bv.v, E0);
    E1 = wmma_f16(A3[1].v, Bv.v, E1);
    E2 = wmma_f16(A3[2].v, Bv.v, E2);
    E3 = wmma_f16(A3[3].v, Bv.v, E3);
#pragma unroll
    for (int i = 0; i < 8; ++i) {
      rmax[0][i] = fmaxf(rmax[0][i], fmaxf(E0[i] + b3[0 * 16 + i + 8 * hf], 0.f));
      rmax[1][i] = fmaxf(rmax[1][i], fmaxf(E1[i] + b3[1 * 16 + i + 8 * hf], 0.f));
      rmax[2][i] = fmaxf(rmax[2][i], fmaxf(E2[i] + b3[2 * 16 + i + 8 * hf], 0.f));
      rmax[3][i] = fmaxf(rmax[3][i], fmaxf(E3[i] + b3[3 * 16 + i + 8 * hf], 0.f));
    }
  }

  // Max over the 16 columns within each lane half.
#pragma unroll
  for (int t = 0; t < 4; ++t)
#pragma unroll
    for (int i = 0; i < 8; ++i) {
      float v = rmax[t][i];
#pragma unroll
      for (int off = 1; off < 16; off <<= 1) v = fmaxf(v, __shfl_xor(v, off));
      rmax[t][i] = v;
    }

  const int cf = cnt[wave * 2 + g] > 0;
  if (l15 == 0) {  // lanes 0 (rows i) and 16 (rows i+8)
#pragma unroll
    for (int t = 0; t < 4; ++t)
#pragma unroll
      for (int i = 0; i < 8; ++i) {
        int row = t * 16 + i + 8 * hf;
        outfeat[((size_t)b * 128 + (choff + row)) * NPT + j] =
            cf ? rmax[t][i] : 0.f;
      }
  }
}

// ---------------------------------------------------------------------------
// Kernel 4: aggregation 128->128 GEMM + ReLU, fused 128->1 confidence head.
// One wave per 16-pixel tile; weights staged in LDS as f16 (row-major, so
// A-operand builds are two contiguous b128 DS loads via the a_kmap runs).
// ---------------------------------------------------------------------------
__global__ __launch_bounds__(256) void agg_kernel(
    const float* __restrict__ feat, const float* __restrict__ aw,
    const float* __restrict__ ab, const float* __restrict__ cw,
    const float* __restrict__ cb, float* __restrict__ nf,
    float* __restrict__ scores) {
  __shared__ _Float16 W[AGG * AGG];      // 32 KB, row-major [o][c]
  __shared__ _Float16 xb[8][16 * AGG];   // 32 KB, column-major [col][c]
  for (int i = threadIdx.x; i < AGG * AGG; i += 256) W[i] = (_Float16)aw[i];
  __syncthreads();

  const int wave = (blockIdx.x * blockDim.x + threadIdx.x) >> 5;  // 0..1023
  const int lane = threadIdx.x & 31;
  const int hf   = lane >> 4;
  const int l15  = lane & 15;
  const int wlb  = (threadIdx.x >> 5) & 7;
  const int b    = wave >> 8;            // 256 tiles per batch
  const int j0   = (wave & 255) * 16;

  if (lane < 16) {
#pragma unroll
    for (int cc = 0; cc < 8; ++cc) {
      V16H tbuf;
#pragma unroll
      for (int c = 0; c < 16; ++c)
        tbuf.h[c] =
            (_Float16)feat[((size_t)b * AGG + cc * 16 + c) * NPT + j0 + lane];
      *(v16h*)&xb[wlb][lane * AGG + cc * 16] = tbuf.v;  // 2x ds_store_b128
    }
  }
  LDS_FENCE();

  V16H Bv[4];
#pragma unroll
  for (int kk = 0; kk < 4; ++kk)
    Bv[kk].v = *(const v16h*)&xb[wlb][l15 * AGG + kk * 32 + 16 * hf];

  float cpart = 0.f;
  for (int co = 0; co < 8; ++co) {
    v8f acc = {};
#pragma unroll
    for (int kk = 0; kk < 4; ++kk) {
      V16H Av;  // two contiguous 8-half runs of W row (a_kmap layout)
      Av.q[0] = *(const v8h*)&W[(co * 16 + l15) * AGG + kk * 32 + 16 * hf];
      Av.q[1] = *(const v8h*)&W[(co * 16 + l15) * AGG + kk * 32 + 16 * hf + 8];
      acc = wmma_f16(Av.v, Bv[kk].v, acc);
    }
#pragma unroll
    for (int i = 0; i < 8; ++i) {
      int   row = co * 16 + i + 8 * hf;
      float v   = fmaxf(acc[i] + ab[row], 0.f);
      nf[((size_t)b * AGG + row) * NPT + j0 + l15] = v;
      cpart += cw[row] * v;
    }
  }
  cpart += __shfl_xor(cpart, 16);
  if (lane < 16) scores[b * NPT + j0 + lane] = cpart + cb[0];
}

// ---------------------------------------------------------------------------
// Host-side launcher.
// d_in leaf order: xyz, features, mlps[0](w,b x3), mlps[1](w,b x3),
//                  agg_w, agg_b, conf_w, conf_b
// d_out: new_xyz (4*4096*3) | nf (4*128*4096) | scores (4*4096)
// ---------------------------------------------------------------------------
extern "C" void kernel_launch(void* const* d_in, const int* in_sizes, int n_in,
                              void* d_out, int out_size, void* d_ws,
                              size_t ws_size, hipStream_t stream) {
  (void)in_sizes; (void)n_in; (void)out_size; (void)ws_size;

  const float* xyz = (const float*)d_in[0];
  const float* fts = (const float*)d_in[1];
  const float* w10 = (const float*)d_in[2];
  const float* b10 = (const float*)d_in[3];
  const float* w11 = (const float*)d_in[4];
  const float* b11 = (const float*)d_in[5];
  const float* w12 = (const float*)d_in[6];
  const float* b12 = (const float*)d_in[7];
  const float* w20 = (const float*)d_in[8];
  const float* b20 = (const float*)d_in[9];
  const float* w21 = (const float*)d_in[10];
  const float* b21 = (const float*)d_in[11];
  const float* w22 = (const float*)d_in[12];
  const float* b22 = (const float*)d_in[13];
  const float* agw = (const float*)d_in[14];
  const float* agb = (const float*)d_in[15];
  const float* cfw = (const float*)d_in[16];
  const float* cfb = (const float*)d_in[17];

  float* out    = (float*)d_out;
  float* newxyz = out;
  float* nf     = out + (size_t)BB * NPT * 3;
  float* scores = out + (size_t)BB * NPT * 3 + (size_t)BB * AGG * NPT;

  // Workspace layout (bytes)
  char*  ws   = (char*)d_ws;
  int*   sidx = (int*)(ws);                                 // 64 KB
  int*   idx0 = (int*)(ws + (1u << 16));                    // 1 MB
  int*   idx1 = (int*)(ws + (1u << 16) + (1u << 20));       // 2 MB
  int*   cnt  = (int*)(ws + (1u << 16) + 3u * (1u << 20));  // 128 KB
  float* fcat = (float*)(ws + (1u << 16) + 3u * (1u << 20) + (1u << 18));
  // fcat: 4*128*4096 floats (8 MB); total ws need ~11.4 MB

  fps_kernel<<<BB, 1024, 0, stream>>>(xyz, sidx, newxyz);

  const int waves  = BB * NPT;   // 16384 groups / samples
  const int blocks = waves / 8;  // 8 waves per 256-thread block
  ballq_kernel<<<blocks, 256, 0, stream>>>(xyz, newxyz, idx0, idx1, cnt);

  mlp_kernel<<<blocks, 256, 0, stream>>>(xyz, fts, newxyz, idx0, cnt, 0, 16,
                                         w10, b10, w11, b11, w12, b12, fcat, 0);
  mlp_kernel<<<blocks, 256, 0, stream>>>(xyz, fts, newxyz, idx1, cnt, 1, 32,
                                         w20, b20, w21, b21, w22, b22, fcat, 64);

  agg_kernel<<<(BB * NPT / 16) / 8, 256, 0, stream>>>(fcat, agw, agb, cfw, cfb,
                                                      nf, scores);
}
